// mp_conv_v2_56495999811914
// MI455X (gfx1250) — compile-verified, matrix operand order checked by reading, not run on hardware
//
#include <hip/hip_runtime.h>

// ---------------------------------------------------------------------------
// Fused mp_conv_v2 for MI455X (gfx1250, wave32, WMMA bf16 16x16x32).
// One wave per (b,n) node: its 16 neighbors == the WMMA N dimension.
// All channel contractions -> v_wmma_f32_16x16x32_bf16, activations live in
// a per-wave LDS arena (px-major), weights prepacked to bf16 (BN folded).
// ---------------------------------------------------------------------------

typedef __attribute__((ext_vector_type(16))) __bf16 v16bf;
typedef __attribute__((ext_vector_type(8)))  float  v8f;

#define BN_EPS 1e-5f

struct KParams {
    const float* node_feature;   // [8,64,1024,1]
    const float* pair_weight;    // [8,16,1024,16]
    const float* etype;          // [8,4,1024,16]
    const float* nfilter;        // [64,512] f32 (kept f32)
    const float* bias;           // [128]
    const float* bng;            // [128]
    const float* bnb;            // [128]
    const int*   nn_idx;         // [8,1024,16]
    // bf16 prepacked weights (row-major [cout][cinP])
    const __bf16 *wN1,*wN2;      // wdy_node  96x128 , 192x96
    const __bf16 *wP1,*wP2;      // wdy_plus  96x32(pad) , 192x96
    const __bf16 *wQ1,*wQ2;      // wdy_pure  96x32(pad) , 192x96
    const __bf16 *wM1,*wM2;      // wdy_mix   128x192 , 128x128
    const __bf16 *wE1,*wE2;      // node      64x128 , 128x64
    const __bf16 *wW1,*wW2;      // weight    64x32(pad) , 128x64
    const __bf16 *wfT;           // wfilter_node^T  128x128
    const __bf16 *F2;            // filters2 as 4 x [128][128], F2e[d][c]
    // f32 folded biases (+ zero bias for wfT layer)
    const float *bN1,*bN2,*bP1,*bP2,*bQ1,*bQ2,*bM1,*bM2,*bE1,*bE2,*bW1,*bW2,*bZ;
    float* out;                  // [8,128,1024,1]
    float* fw;                   // [8,128,1024,16]
};

// bf16 bit helpers
__device__ inline float bfu_lo(unsigned u) {
    union { float f; unsigned u; } x; x.u = u << 16;        return x.f;
}
__device__ inline float bfu_hi(unsigned u) {
    union { float f; unsigned u; } x; x.u = u & 0xffff0000u; return x.f;
}
__device__ inline unsigned pack_bf2(float a, float b) {
    __bf16 p0 = (__bf16)a, p1 = (__bf16)b;
    return ((unsigned)__builtin_bit_cast(unsigned short, p1) << 16) |
           (unsigned)__builtin_bit_cast(unsigned short, p0);
}

// ---- fragment loaders (bf16 16x16x32: lanes 0-15 hold K {0..7,16..23},
// ---- lanes 16-31 hold K {8..15,24..31}; A row = lane&15, B col = lane&15) ----
__device__ inline v16bf load_w_frag(const __bf16* W, int cinP, int m0, int k0, int lane) {
    const int m = m0 + (lane & 15), hi = (lane >> 4) & 1;
    const __bf16* row = W + m * cinP + k0 + 8 * hi;
    v16bf a;
#pragma unroll
    for (int i = 0; i < 16; ++i) a[i] = row[(i & 7) + 16 * (i >> 3)];
    return a;
}
__device__ inline v16bf load_b_frag(const __bf16* X, int xs, int k0, int lane) {
    const int n = lane & 15, hi = (lane >> 4) & 1;
    const __bf16* col = X + n * xs + k0 + 8 * hi;
    v16bf b;
#pragma unroll
    for (int i = 0; i < 16; ++i) b[i] = col[(i & 7) + 16 * (i >> 3)];
    return b;
}
__device__ inline v8f wmma_bf16(v16bf a, v16bf b, v8f c) {
    return __builtin_amdgcn_wmma_f32_16x16x32_bf16(false, a, false, b, (short)0, c,
                                                   false, false);
}
__device__ inline v8f gemm_frag(const __bf16* W, int cinP, int m0,
                                const __bf16* X, int xs, int lane) {
    v8f c = {};
    for (int k0 = 0; k0 < cinP; k0 += 32)
        c = wmma_bf16(load_w_frag(W, cinP, m0, k0, lane),
                      load_b_frag(X, xs, k0, lane), c);
    return c;
}
// Y[px][ch] = (relu?)(W X + b); paired bf16 stores (ch, ch+1 contiguous, aligned)
template <bool RELU>
__device__ inline void layer(const __bf16* W, const float* bias, int cout, int cinP,
                             const __bf16* X, int xs, __bf16* Y, int ys, int lane) {
    const int px = lane & 15, hi = (lane >> 4) & 1;
    for (int m0 = 0; m0 < cout; m0 += 16) {
        v8f c = gemm_frag(W, cinP, m0, X, xs, lane);
#pragma unroll
        for (int r = 0; r < 8; r += 2) {
            const int ch = m0 + r + 8 * hi;
            float v0 = c[r]     + bias[ch];
            float v1 = c[r + 1] + bias[ch + 1];
            if (RELU) { v0 = fmaxf(v0, 0.f); v1 = fmaxf(v1, 0.f); }
            *(unsigned*)&Y[px * ys + ch] = pack_bf2(v0, v1);
        }
    }
}

#define WPW 2                      // waves per workgroup
constexpr int ARENA = 12288;       // bf16 per wave (24 KB)

__global__ __launch_bounds__(WPW * 32) void fused_main(KParams P) {
    __shared__ __bf16 smem[WPW * ARENA];
    __shared__ float  nftb[WPW * 512];
    const int lane = threadIdx.x & 31;
    const int wv   = threadIdx.x >> 5;
    const int gw   = blockIdx.x * WPW + wv;       // 0..8191
    const int b = gw >> 10, n = gw & 1023;
    __bf16* A   = smem + wv * ARENA;
    float*  NFT = nftb + wv * 512;
    __bf16 *PNF = A,            *PWP = A + 2048, *T96  = A + 2560,
           *NFE = A + 4096,     *MIX = A + 7168, *T128 = A + 10240;
    const int px = lane & 15, hi = (lane >> 4) & 1;

    // phase 1: gather pnf (self || neighbor) and padded pair_weight tile
    for (int i = lane; i < 2048; i += 32) {
        const int p = i >> 7, ch = i & 127;
        float v;
        if (ch < 64) v = P.node_feature[(b * 64 + ch) * 1024 + n];
        else {
            const int m = P.nn_idx[(b * 1024 + n) * 16 + p];
            v = P.node_feature[(b * 64 + (ch - 64)) * 1024 + m];
        }
        PNF[p * 128 + ch] = (__bf16)v;
    }
    for (int i = lane; i < 512; i += 32) {
        const int p = i >> 5, ch = i & 31;
        const float v = (ch < 16) ? P.pair_weight[((b * 16 + ch) * 1024 + n) * 16 + p] : 0.f;
        PWP[p * 32 + ch] = (__bf16)v;
    }
    __syncthreads();

    // wdy_node: pnf -> 96(relu) -> 192 (nfe)
    layer<true >(P.wN1, P.bN1, 96, 128, PNF, 128, T96, 96,  lane); __syncthreads();
    layer<false>(P.wN2, P.bN2, 192, 96, T96,  96, NFE, 192, lane); __syncthreads();
    // wdy_pure: pw -> 96(relu) -> 192 (wog, accumulated into MIX)
    layer<true >(P.wQ1, P.bQ1, 96,  32, PWP,  32, T96, 96,  lane); __syncthreads();
    layer<false>(P.wQ2, P.bQ2, 192, 96, T96,  96, MIX, 192, lane); __syncthreads();
    // wdy_plus: pw -> 96(relu); then MIX += NFE * (W2 t + b2)
    layer<true >(P.wP1, P.bP1, 96,  32, PWP,  32, T96, 96,  lane); __syncthreads();
    for (int m0 = 0; m0 < 192; m0 += 16) {
        v8f c = gemm_frag(P.wP2, 96, m0, T96, 96, lane);
#pragma unroll
        for (int r = 0; r < 8; r += 2) {
            const int ch = m0 + r + 8 * hi;
            const unsigned um = *(const unsigned*)&MIX[px * 192 + ch];
            const unsigned uf = *(const unsigned*)&NFE[px * 192 + ch];
            const float w0 = c[r]     + P.bP2[ch];
            const float w1 = c[r + 1] + P.bP2[ch + 1];
            const float m0v = bfu_lo(um) + bfu_lo(uf) * w0;
            const float m1v = bfu_hi(um) + bfu_hi(uf) * w1;
            *(unsigned*)&MIX[px * 192 + ch] = pack_bf2(m0v, m1v);
        }
    }
    __syncthreads();
    // wdy_mix: MIX -> 128(relu) -> final_weight (global f32)
    layer<true>(P.wM1, P.bM1, 128, 192, MIX, 192, T128, 128, lane); __syncthreads();
    for (int m0 = 0; m0 < 128; m0 += 16) {
        v8f c = gemm_frag(P.wM2, 128, m0, T128, 128, lane);
#pragma unroll
        for (int r = 0; r < 8; ++r) {
            const int ch = m0 + r + 8 * hi;
            P.fw[((b * 128 + ch) * 1024 + n) * 16 + px] = c[r] + P.bM2[ch];
        }
    }
    __syncthreads();

    // main path: h = seq(pnf; node), pwv = seq(pw; weight), med = wfT * pwv
    __bf16 *H = NFE, *PWV = MIX;  // reuse dead regions
    layer<true >(P.wE1, P.bE1, 64, 128, PNF, 128, T96, 64,  lane); __syncthreads();
    layer<false>(P.wE2, P.bE2, 128, 64, T96,  64, H,  128,  lane); __syncthreads();
    layer<true >(P.wW1, P.bW1, 64,  32, PWP,  32, T96, 64,  lane); __syncthreads();
    layer<false>(P.wW2, P.bW2, 128, 64, T96,  64, PWV, 128, lane); __syncthreads();
    layer<false>(P.wfT, P.bZ, 128, 128, PWV, 128, T128, 128, lane); __syncthreads();

    // h *= med ; nft[d,e] = sum_c nf_self[c] * nfilter[c, d*4+e]
    for (int i = lane; i < 1024; i += 32) {   // paired: 2048 bf16 as 1024 u32
        const unsigned uh = ((const unsigned*)H)[i];
        const unsigned ut = ((const unsigned*)T128)[i];
        ((unsigned*)H)[i] = pack_bf2(bfu_lo(uh) * bfu_lo(ut), bfu_hi(uh) * bfu_hi(ut));
    }
    for (int j = 0; j < 16; ++j) {
        const int idx = lane * 16 + j;            // == d*4+e
        float s = 0.f;
        for (int c = 0; c < 64; ++c) s += (float)PNF[c] * P.nfilter[c * 512 + idx];
        NFT[idx] = s;
    }
    __syncthreads();

    // edge = sum_e (F2e^T h + nft) * etype[e]; max over k; +bias; BN; ReLU
    float etv[4];
#pragma unroll
    for (int e = 0; e < 4; ++e) etv[e] = P.etype[((b * 4 + e) * 1024 + n) * 16 + px];
    const float bns = rsqrtf(1.f + BN_EPS);
    for (int m0 = 0; m0 < 128; m0 += 16) {
        float acc[8];
#pragma unroll
        for (int r = 0; r < 8; ++r) {
            const int ch = m0 + r + 8 * hi;
            float s = 0.f;
#pragma unroll
            for (int e = 0; e < 4; ++e) s += NFT[ch * 4 + e] * etv[e];
            acc[r] = s;
        }
        for (int e = 0; e < 4; ++e) {
            v8f c = gemm_frag(P.F2 + e * 16384, 128, m0, H, 128, lane);
#pragma unroll
            for (int r = 0; r < 8; ++r) acc[r] += c[r] * etv[e];
        }
#pragma unroll
        for (int r = 0; r < 8; ++r) {
            float v = acc[r];
            for (int off = 1; off < 16; off <<= 1)       // max over k == 16 lanes
                v = fmaxf(v, __shfl_xor(v, off, 32));
            const int ch = m0 + r + 8 * hi;
            v += P.bias[ch];
            v = v * (P.bng[ch] * bns) + P.bnb[ch];
            v = fmaxf(v, 0.f);
            if ((lane & 15) == 0) P.out[(b * 128 + ch) * 1024 + n] = v;
        }
    }
}

// ---- weight prep: BN-fold conv1, convert to bf16, pad cin 16->32 ----
__global__ void prep_seq(const float* w1, const float* b1, const float* g1,
                         const float* be1, const float* w2, const float* b2,
                         __bf16* W1, float* B1, __bf16* W2, float* B2,
                         int cmid, int cin, int cinP, int cout) {
    const int tid = blockIdx.x * blockDim.x + threadIdx.x;
    const int st  = gridDim.x * blockDim.x;
    const float is = rsqrtf(1.f + BN_EPS);
    for (int i = tid; i < cmid * cinP; i += st) {
        const int o = i / cinP, c = i % cinP;
        const float s = g1[o] * is;
        W1[i] = (__bf16)((c < cin) ? w1[o * cin + c] * s : 0.f);
    }
    for (int i = tid; i < cmid; i += st) B1[i] = b1[i] * g1[i] * is + be1[i];
    for (int i = tid; i < cout * cmid; i += st) W2[i] = (__bf16)w2[i];
    for (int i = tid; i < cout; i += st) B2[i] = b2[i];
}

__global__ void prep_misc(const float* wf, const float* f2, __bf16* wfT, __bf16* F2,
                          float* zb) {
    const int tid = blockIdx.x * blockDim.x + threadIdx.x;
    const int st  = gridDim.x * blockDim.x;
    for (int i = tid; i < 128 * 128; i += st) {        // wfT[d][w] = wf[w][d]
        const int d = i >> 7, w = i & 127;
        wfT[i] = (__bf16)wf[w * 128 + d];
    }
    for (int i = tid; i < 4 * 128 * 128; i += st) {    // F2e[d][c] = filters2[c,d,e]
        const int e = i >> 14, r = i & 16383, d = r >> 7, c = r & 127;
        F2[i] = (__bf16)f2[(c * 128 + d) * 4 + e];
    }
    for (int i = tid; i < 128; i += st) zb[i] = 0.f;   // zero bias
}

extern "C" void kernel_launch(void* const* d_in, const int* in_sizes, int n_in,
                              void* d_out, int out_size, void* d_ws, size_t ws_size,
                              hipStream_t stream) {
    auto f = [&](int i) { return (const float*)d_in[i]; };
    __bf16* wsb = (__bf16*)d_ws;
    float*  wsf = (float*)((char*)d_ws + 446464);      // after 223232 bf16 elems

    dim3 pb(256), pg(64);
    // inputs: 0 nf, 1 pw, 2 nn_idx, 3 etype, 4-9 wdy_node, 10-15 wdy_plus,
    // 16-21 wdy_pure, 22-27 wdy_mix, 28-33 node, 34-39 weight, 40 wf, 41 nfilter,
    // 42 filters2, 43 bias, 44 bn_g, 45 bn_b
    prep_seq<<<pg, pb, 0, stream>>>(f(4),  f(5),  f(6),  f(7),  f(8),  f(9),
        wsb + 0,      wsf + 0,    wsb + 12288,  wsf + 96,   96, 128, 128, 192);
    prep_seq<<<pg, pb, 0, stream>>>(f(10), f(11), f(12), f(13), f(14), f(15),
        wsb + 30720,  wsf + 288,  wsb + 33792,  wsf + 384,  96, 16,  32,  192);
    prep_seq<<<pg, pb, 0, stream>>>(f(16), f(17), f(18), f(19), f(20), f(21),
        wsb + 52224,  wsf + 576,  wsb + 55296,  wsf + 768,  96, 16,  32,  192);
    prep_seq<<<pg, pb, 0, stream>>>(f(22), f(23), f(24), f(25), f(26), f(27),
        wsb + 73728,  wsf + 960,  wsb + 98304,  wsf + 1088, 128, 192, 192, 128);
    prep_seq<<<pg, pb, 0, stream>>>(f(28), f(29), f(30), f(31), f(32), f(33),
        wsb + 114688, wsf + 1216, wsb + 122880, wsf + 1280, 64, 128, 128, 128);
    prep_seq<<<pg, pb, 0, stream>>>(f(34), f(35), f(36), f(37), f(38), f(39),
        wsb + 131072, wsf + 1408, wsb + 133120, wsf + 1472, 64, 16,  32,  128);
    prep_misc<<<pg, pb, 0, stream>>>(f(40), f(42), wsb + 141312, wsb + 157696,
                                     wsf + 1600);

    KParams P;
    P.node_feature = f(0); P.pair_weight = f(1); P.etype = f(3);
    P.nfilter = f(41); P.bias = f(43); P.bng = f(44); P.bnb = f(45);
    P.nn_idx = (const int*)d_in[2];
    P.wN1 = wsb + 0;      P.wN2 = wsb + 12288;
    P.wP1 = wsb + 30720;  P.wP2 = wsb + 33792;
    P.wQ1 = wsb + 52224;  P.wQ2 = wsb + 55296;
    P.wM1 = wsb + 73728;  P.wM2 = wsb + 98304;
    P.wE1 = wsb + 114688; P.wE2 = wsb + 122880;
    P.wW1 = wsb + 131072; P.wW2 = wsb + 133120;
    P.wfT = wsb + 141312; P.F2  = wsb + 157696;
    P.bN1 = wsf + 0;    P.bN2 = wsf + 96;
    P.bP1 = wsf + 288;  P.bP2 = wsf + 384;
    P.bQ1 = wsf + 576;  P.bQ2 = wsf + 768;
    P.bM1 = wsf + 960;  P.bM2 = wsf + 1088;
    P.bE1 = wsf + 1216; P.bE2 = wsf + 1280;
    P.bW1 = wsf + 1408; P.bW2 = wsf + 1472;
    P.bZ  = wsf + 1600;
    P.out = (float*)d_out;
    P.fw  = (float*)d_out + 1048576;

    fused_main<<<4096, WPW * 32, 0, stream>>>(P);
}
